// Attention_13649406067265
// MI455X (gfx1250) — matrix operand order repaired and checked by reading, not compiled, not run on hardware
//
#include <hip/hip_runtime.h>
#include <hip/hip_bf16.h>

typedef __attribute__((ext_vector_type(16))) _Float16 v16h;
typedef __attribute__((ext_vector_type(8)))  _Float16 v8h;
typedef __attribute__((ext_vector_type(8)))  float    v8f;

#define DIMC    512
#define NHEADS  16
#define HD      32
#define BATCH   8
#define SEQ     1024
#define TABLE_N 3969
#define SEQ_SCALE 6.9314718055994531f   // log(32*32)

static __device__ __forceinline__ v8f wmma_f16(v16h a, v16h b, v8f c) {
    // D = A(16x32 f16) x B(32x16 f16) + C(16x16 f32)
    return __builtin_amdgcn_wmma_f32_16x16x32_f16(false, a, false, b, (short)0, c, false, false);
}

// A-operand load: lane holds row (lane&15); element e -> K = 16*hi + e (16 contiguous)
static __device__ __forceinline__ v16h load_a16(const _Float16* __restrict__ rowbase, int hi) {
    v8h lo = *(const v8h*)(rowbase + 16 * hi);
    v8h hv = *(const v8h*)(rowbase + 16 * hi + 8);
    v16h r;
#pragma unroll
    for (int e = 0; e < 8; ++e) { r[e] = lo[e]; r[8 + e] = hv[e]; }
    return r;
}

// B-operand load: lane holds col (lane&15); element e -> K = 16*hi + e (16 contiguous)
static __device__ __forceinline__ v16h load_b16(const _Float16* __restrict__ colbase, int hi) {
    v8h lo = *(const v8h*)(colbase + 16 * hi);
    v8h hv = *(const v8h*)(colbase + 16 * hi + 8);
    v16h r;
#pragma unroll
    for (int e = 0; e < 8; ++e) { r[e] = lo[e]; r[8 + e] = hv[e]; }
    return r;
}

// ---------------------------------------------------------------- f32 -> f16
__global__ void cvt_kernel(const float* __restrict__ src, _Float16* __restrict__ dst, int n) {
    int i = blockIdx.x * blockDim.x + threadIdx.x;
    int stride = gridDim.x * blockDim.x;
    for (; i < n; i += stride) dst[i] = (_Float16)src[i];
}

// ---------------------------------------------------------------- CPB MLP: [T,2] -> relu -> [T,512] -> [T,16]
__global__ void cpb_kernel(const float* __restrict__ table,
                           const float* __restrict__ fc1w, const float* __restrict__ fc1b,
                           const float* __restrict__ fc2w, const float* __restrict__ fc2b,
                           float* __restrict__ tbl) {
    __shared__ float hid[512];
    const int t = blockIdx.x;
    const float c0 = table[t * 2 + 0];
    const float c1 = table[t * 2 + 1];
    for (int j = threadIdx.x; j < 512; j += blockDim.x) {
        float v = c0 * fc1w[j * 2 + 0] + c1 * fc1w[j * 2 + 1] + fc1b[j];
        hid[j] = fmaxf(v, 0.f);
    }
    __syncthreads();
    if (threadIdx.x < NHEADS) {
        const int hh = threadIdx.x;
        float s = fc2b[hh];
        for (int j = 0; j < 512; ++j) s += hid[j] * fc2w[hh * 512 + j];
        tbl[t * NHEADS + hh] = s;
    }
}

// ---------------------------------------------------------------- rpi transpose (LDS tile) for coalesced expand
__global__ void rpi_transpose_kernel(const int* __restrict__ rpi, int* __restrict__ rpiT) {
    __shared__ int tile[32][33];
    const int bx = blockIdx.x * 32, by = blockIdx.y * 32;
    const int tx = threadIdx.x & 31, ty = threadIdx.x >> 5;   // 32x8
#pragma unroll
    for (int i = 0; i < 32; i += 8)
        tile[ty + i][tx] = rpi[(size_t)(by + ty + i) * SEQ + bx + tx];
    __syncthreads();
#pragma unroll
    for (int i = 0; i < 32; i += 8)
        rpiT[(size_t)(bx + ty + i) * SEQ + by + tx] = tile[tx][ty + i];
}

// ---------------------------------------------------------------- expand bias: biasT[h][k][q] = tbl[rpi[q][k]][h]
__global__ void bias_expand_kernel(const int* __restrict__ rpiT, const float* __restrict__ tbl,
                                   float* __restrict__ biasT) {
    const int idx = blockIdx.x * blockDim.x + threadIdx.x;    // k*SEQ + q  (q fastest -> coalesced)
    const int t = rpiT[idx];                                  // coalesced read
    const float4* t4 = (const float4*)(tbl + (size_t)t * NHEADS);   // one 64B table row
    float vals[16];
    *(float4*)&vals[0]  = t4[0];
    *(float4*)&vals[4]  = t4[1];
    *(float4*)&vals[8]  = t4[2];
    *(float4*)&vals[12] = t4[3];
#pragma unroll
    for (int h = 0; h < NHEADS; ++h)
        biasT[(size_t)h * SEQ * SEQ + idx] = vals[h];         // 16 coalesced streams
}

// ---------------------------------------------------------------- QKV GEMM (WMMA) with scatter epilogue
// C3[8192,1536] = xh @ wqkv^T + b  ->  q/k/v laid out [B,H,N,32] f32
__global__ void qkv_gemm_kernel(const _Float16* __restrict__ xh,
                                const _Float16* __restrict__ wh,
                                const float* __restrict__ qkvb,
                                float* __restrict__ qb, float* __restrict__ kb, float* __restrict__ vb) {
    const int lane = threadIdx.x & 31;
    const int wave = threadIdx.x >> 5;
    const int hi = lane >> 4;
    const int lo = lane & 15;
    const int mtile = blockIdx.x * 128 + wave * 16;
    const int ctile = blockIdx.y * 32;

    const _Float16* arow  = xh + (size_t)(mtile + lo) * DIMC;
    const _Float16* brow1 = wh + (size_t)(ctile + lo) * DIMC;
    const _Float16* brow2 = wh + (size_t)(ctile + 16 + lo) * DIMC;

    v8f acc1 = {}, acc2 = {};
#pragma unroll 4
    for (int kk = 0; kk < DIMC; kk += 32) {
        v16h a  = load_a16(arow + kk, hi);
        v16h b1 = load_b16(brow1 + kk, hi);
        v16h b2 = load_b16(brow2 + kk, hi);
        acc1 = wmma_f16(a, b1, acc1);
        acc2 = wmma_f16(a, b2, acc2);
    }
    const int c1 = ctile + lo, c2 = c1 + 16;
    const float bias1 = qkvb[c1], bias2 = qkvb[c2];
    const int s  = ctile >> 9;            // 0=q 1=k 2=v (32-col tile never crosses a head boundary)
    const int hh = (ctile >> 5) & 15;
    float* dst = (s == 0) ? qb : ((s == 1) ? kb : vb);
#pragma unroll
    for (int r = 0; r < 8; ++r) {
        const int tok = mtile + r + 8 * hi;      // D row = token
        const int bi = tok >> 10, n = tok & 1023;
        float* p = dst + (((size_t)(bi * NHEADS + hh)) * SEQ + n) * HD;
        p[lo]      = acc1[r] + bias1;
        p[16 + lo] = acc2[r] + bias2;
    }
}

// ---------------------------------------------------------------- L2-normalize q,k; scale; build Vt
__global__ void norm_kernel(const float* __restrict__ qb, const float* __restrict__ kb,
                            const float* __restrict__ vb,
                            const float* __restrict__ temperature, const float* __restrict__ qemb,
                            _Float16* __restrict__ qn, _Float16* __restrict__ kn,
                            _Float16* __restrict__ vT) {
    const int lane = threadIdx.x & 31;
    const int warp = threadIdx.x >> 5;
    const int row = blockIdx.x * 8 + warp;       // (b*16+h)*1024 + n
    const int bh = row >> 10;
    const int n  = row & 1023;
    const int h  = bh & 15;
    const int d  = lane;

    const float qv = qb[(size_t)row * HD + d];
    const float kv = kb[(size_t)row * HD + d];
    const float vv = vb[(size_t)row * HD + d];

    float qs = qv * qv, ks = kv * kv;
#pragma unroll
    for (int off = 16; off >= 1; off >>= 1) {
        qs += __shfl_xor(qs, off);
        ks += __shfl_xor(ks, off);
    }
    const float qnrm = fmaxf(sqrtf(qs), 1e-12f);
    const float knrm = fmaxf(sqrtf(ks), 1e-12f);
    const float scale = log1pf(__expf(temperature[h])) * SEQ_SCALE;   // softplus * seq_scale
    qn[(size_t)row * HD + d] = (_Float16)((qv / qnrm + qemb[h * HD + d]) * scale);
    kn[(size_t)row * HD + d] = (_Float16)(kv / knrm);
    vT[((size_t)bh * HD + d) * SEQ + n] = (_Float16)vv;               // transposed for PV A-operand
}

// ---------------------------------------------------------------- fused flash attention (transposed form)
// S' = kn . qn^T  (one WMMA per 16x16, K = hd = 32);  O^T = V^T . P^T
__global__ void attn_kernel(const _Float16* __restrict__ qn,
                            const _Float16* __restrict__ kn,
                            const _Float16* __restrict__ vT,
                            const float* __restrict__ biasT,
                            _Float16* __restrict__ xat) {
    const int lane = threadIdx.x & 31;
    const int wave = threadIdx.x >> 5;
    const int hi = lane >> 4;
    const int lo = lane & 15;
    const int bh = blockIdx.x;                   // b*16 + h
    const int h  = bh & 15;
    const int qbase = blockIdx.y * 128 + wave * 16;
    const int q = qbase + lo;                    // this lane's query column (lane-local stats!)

    const v16h bq = load_b16(qn + ((size_t)bh * SEQ + q) * HD, hi);   // loop-invariant B operand
    const float* __restrict__ brow = biasT + (size_t)h * SEQ * SEQ + q;   // coalesced bias stream
    const _Float16* __restrict__ knb = kn + (size_t)bh * SEQ * HD;
    const _Float16* __restrict__ vb1 = vT + ((size_t)bh * HD + lo) * SEQ;
    const _Float16* __restrict__ vb2 = vb1 + (size_t)16 * SEQ;

    v8f o1 = {}, o2 = {};
    const v8f zero = {};
    float m_run = -1e30f, l_run = 0.f;

    for (int kt = 0; kt < SEQ; kt += 32) {
        // prefetch next K-tile of the two sequential streams
        if (kt + 32 < SEQ) {
            __builtin_prefetch(knb + (size_t)(kt + 32 + lo) * HD, 0, 1);
            __builtin_prefetch(vb1 + kt + 32, 0, 1);
        }
        // scores: two 16(k) x 16(q) tiles
        v16h ka1 = load_a16(knb + (size_t)(kt + lo) * HD, hi);
        v16h ka2 = load_a16(knb + (size_t)(kt + 16 + lo) * HD, hi);
        v8f s1 = wmma_f16(ka1, bq, zero);
        v8f s2 = wmma_f16(ka2, bq, zero);

        // + relative-position bias (coalesced: lanes vary q)
        float sv1[8], sv2[8];
        float mloc = -1e30f;
#pragma unroll
        for (int r = 0; r < 8; ++r) {
            const int k1 = kt + r + 8 * hi;      // D row = key index
            sv1[r] = s1[r] + brow[(size_t)k1 * SEQ];
            sv2[r] = s2[r] + brow[(size_t)(k1 + 16) * SEQ];
            mloc = fmaxf(mloc, fmaxf(sv1[r], sv2[r]));
        }
        // online softmax over k; q is lane-local so only one cross-half combine needed
        mloc = fmaxf(mloc, __shfl_xor(mloc, 16));
        const float mnew  = fmaxf(m_run, mloc);
        const float alpha = __expf(m_run - mnew);
        float p1[8], p2[8], ssum = 0.f;
#pragma unroll
        for (int r = 0; r < 8; ++r) {
            p1[r] = __expf(sv1[r] - mnew);
            p2[r] = __expf(sv2[r] - mnew);
            ssum += p1[r] + p2[r];
        }
        ssum += __shfl_xor(ssum, 16);
        l_run = l_run * alpha + ssum;
        m_run = mnew;
#pragma unroll
        for (int r = 0; r < 8; ++r) { o1[r] *= alpha; o2[r] *= alpha; }

        // P^T : D-layout -> B-layout via 16 half-wave shuffles (no LDS)
        v16h bp;
#pragma unroll
        for (int e = 0; e < 8; ++e) {
            const float x1 = __shfl_xor(p1[e], 16);   // lanes<16 receive k=8+e
            const float x2 = __shfl_xor(p2[e], 16);   // lanes>=16 receive k=16+e
            bp[e]     = (_Float16)((hi == 0) ? p1[e] : x2);
            bp[8 + e] = (_Float16)((hi == 0) ? x1 : p2[e]);
        }

        // O^T += V^T(hd x 32k) . P^T(32k x 16q)
        v16h va1 = load_a16(vb1 + kt, hi);
        v16h va2 = load_a16(vb2 + kt, hi);
        o1 = wmma_f16(va1, bp, o1);
        o2 = wmma_f16(va2, bp, o2);
    }

    const float inv = 1.0f / l_run;
    _Float16* obase = xat + ((size_t)(bh >> 4) * SEQ + q) * DIMC + h * HD;
    v8h t1, t2;
#pragma unroll
    for (int r = 0; r < 8; ++r) {
        t1[r] = (_Float16)(o1[r] * inv);             // hd = 8*hi + r
        t2[r] = (_Float16)(o2[r] * inv);             // hd = 16 + 8*hi + r
    }
    *(v8h*)(obase + 8 * hi)      = t1;
    *(v8h*)(obase + 16 + 8 * hi) = t2;
}

// ---------------------------------------------------------------- output projection (WMMA) + bias -> f32
__global__ void proj_gemm_kernel(const _Float16* __restrict__ xh,
                                 const _Float16* __restrict__ wh,
                                 const float* __restrict__ pb,
                                 float* __restrict__ out) {
    const int lane = threadIdx.x & 31;
    const int wave = threadIdx.x >> 5;
    const int hi = lane >> 4;
    const int lo = lane & 15;
    const int mtile = blockIdx.x * 128 + wave * 16;
    const int ctile = blockIdx.y * 32;

    const _Float16* arow  = xh + (size_t)(mtile + lo) * DIMC;
    const _Float16* brow1 = wh + (size_t)(ctile + lo) * DIMC;
    const _Float16* brow2 = wh + (size_t)(ctile + 16 + lo) * DIMC;

    v8f acc1 = {}, acc2 = {};
#pragma unroll 4
    for (int kk = 0; kk < DIMC; kk += 32) {
        v16h a  = load_a16(arow + kk, hi);
        v16h b1 = load_b16(brow1 + kk, hi);
        v16h b2 = load_b16(brow2 + kk, hi);
        acc1 = wmma_f16(a, b1, acc1);
        acc2 = wmma_f16(a, b2, acc2);
    }
    const int c1 = ctile + lo, c2 = c1 + 16;
    const float bias1 = pb[c1], bias2 = pb[c2];
#pragma unroll
    for (int r = 0; r < 8; ++r) {
        const int tok = mtile + r + 8 * hi;
        out[(size_t)tok * DIMC + c1] = acc1[r] + bias1;
        out[(size_t)tok * DIMC + c2] = acc2[r] + bias2;
    }
}

// ----------------------------------------------------------------
extern "C" void kernel_launch(void* const* d_in, const int* in_sizes, int n_in,
                              void* d_out, int out_size, void* d_ws, size_t ws_size,
                              hipStream_t stream) {
    (void)in_sizes; (void)n_in; (void)out_size; (void)ws_size;
    const float* x           = (const float*)d_in[0];
    const int*   rpi         = (const int*)d_in[3];
    const float* table       = (const float*)d_in[4];
    const float* qkv_w       = (const float*)d_in[5];
    const float* qkv_b       = (const float*)d_in[6];
    const float* proj_w      = (const float*)d_in[7];
    const float* proj_b      = (const float*)d_in[8];
    const float* temperature = (const float*)d_in[9];
    const float* qemb        = (const float*)d_in[10];
    const float* fc1w        = (const float*)d_in[11];
    const float* fc1b        = (const float*)d_in[12];
    const float* fc2w        = (const float*)d_in[13];
    const float* fc2b        = (const float*)d_in[14];

    char* ws = (char*)d_ws;
    size_t off = 0;
    auto alloc = [&](size_t bytes) -> void* {
        void* p = ws + off;
        off = (off + bytes + 255) & ~(size_t)255;
        return p;
    };
    const size_t TOK = (size_t)BATCH * SEQ;                       // 8192
    _Float16* xh    = (_Float16*)alloc(TOK * DIMC * 2);           // 8 MB
    _Float16* wqkvh = (_Float16*)alloc((size_t)3 * DIMC * DIMC * 2);
    _Float16* wprh  = (_Float16*)alloc((size_t)DIMC * DIMC * 2);
    float*    tbl   = (float*)alloc((size_t)TABLE_N * NHEADS * 4);
    int*      rpiT  = (int*)alloc((size_t)SEQ * SEQ * 4);         // 4 MB
    float*    biasT = (float*)alloc((size_t)NHEADS * SEQ * SEQ * 4); // 64 MB
    float*    qb    = (float*)alloc(TOK * DIMC * 4);              // [B,H,N,32] f32, 16 MB
    float*    kb    = (float*)alloc(TOK * DIMC * 4);
    float*    vb    = (float*)alloc(TOK * DIMC * 4);
    _Float16* qn    = (_Float16*)alloc(TOK * DIMC * 2);
    _Float16* kn    = (_Float16*)alloc(TOK * DIMC * 2);
    _Float16* vT    = (_Float16*)alloc(TOK * DIMC * 2);
    _Float16* xat   = (_Float16*)alloc(TOK * DIMC * 2);

    cvt_kernel<<<1024, 256, 0, stream>>>(x, xh, (int)(TOK * DIMC));
    cvt_kernel<<<512, 256, 0, stream>>>(qkv_w, wqkvh, 3 * DIMC * DIMC);
    cvt_kernel<<<256, 256, 0, stream>>>(proj_w, wprh, DIMC * DIMC);
    cpb_kernel<<<TABLE_N, 256, 0, stream>>>(table, fc1w, fc1b, fc2w, fc2b, tbl);
    rpi_transpose_kernel<<<dim3(SEQ / 32, SEQ / 32), 256, 0, stream>>>(rpi, rpiT);
    bias_expand_kernel<<<SEQ * SEQ / 256, 256, 0, stream>>>(rpiT, tbl, biasT);
    qkv_gemm_kernel<<<dim3(64, 48), 256, 0, stream>>>(xh, wqkvh, qkv_b, qb, kb, vb);
    norm_kernel<<<(int)(BATCH * NHEADS * SEQ / 8), 256, 0, stream>>>(qb, kb, vb, temperature, qemb, qn, kn, vT);
    attn_kernel<<<dim3(BATCH * NHEADS, SEQ / 128), 256, 0, stream>>>(qn, kn, vT, biasT, xat);
    proj_gemm_kernel<<<dim3(64, 16), 256, 0, stream>>>(xat, wprh, proj_b, (float*)d_out);
}
